// UnifiedDeepRanker_82343112999137
// MI455X (gfx1250) — matrix operand order repaired and checked by reading, not compiled, not run on hardware
//
#include <hip/hip_runtime.h>

// ---------------- sizes ----------------
constexpr int BATCH = 16384;
constexpr int SEQT  = 64;
constexpr int DIM   = 128;
constexpr int TOTF  = 784;        // 5*128 + 128 + 16
constexpr int TOTP  = 800;        // TOTF padded to K-step
constexpr int NEXP  = 4;
constexpr int ATTH  = 64;

typedef __attribute__((ext_vector_type(16))) __bf16 v16bf;
typedef __attribute__((ext_vector_type(8)))  float  v8f;
typedef unsigned u32x4 __attribute__((ext_vector_type(4)));
typedef int      i32x8 __attribute__((ext_vector_type(8)));
typedef int      i32x4 __attribute__((ext_vector_type(4)));

static_assert(sizeof(__bf16) == 2, "bf16 size");

#if __has_builtin(__builtin_amdgcn_tensor_load_to_lds) && \
    __has_builtin(__builtin_amdgcn_s_wait_tensorcnt)
#define USE_TDM 1
#else
#define USE_TDM 0
#endif

__device__ __forceinline__ __bf16 f2bf(float f) {
  union { float f; unsigned u; } v; v.f = f;
  unsigned r = v.u + 0x7FFFu + ((v.u >> 16) & 1u);   // RNE
  unsigned short h = (unsigned short)(r >> 16);
  __bf16 out;
  __builtin_memcpy(&out, &h, 2);
  return out;
}

// pack two fp32 -> one dword holding {bf16(a) lo, bf16(b) hi}
__device__ __forceinline__ unsigned packbf(float a, float b) {
  union { float f; unsigned u; } x, y; x.f = a; y.f = b;
  unsigned ra = x.u + 0x7FFFu + ((x.u >> 16) & 1u);
  unsigned rb = y.u + 0x7FFFu + ((y.u >> 16) & 1u);
  return (ra >> 16) | (rb & 0xFFFF0000u);
}

#if USE_TDM
// 2D TDM: copy [nRows x 32 bf16] tile (row pitch = pitchElems) from global
// into LDS at ldsByte, LDS row pitch 80B (64B data + 16B pad).
__device__ __forceinline__ void tdm_tile_2d(const __bf16* gsrc, unsigned ldsByte,
                                            int nRows, int pitchElems)
{
  unsigned long long ga = (unsigned long long)(const char*)gsrc;
  u32x4 g0;
  g0[0] = 1u;                                               // count=1, user mode
  g0[1] = ldsByte;                                          // lds_addr
  g0[2] = (unsigned)(ga & 0xFFFFFFFFu);                     // global_addr[31:0]
  g0[3] = (unsigned)((ga >> 32) & 0x1FFFFFFu) | (2u << 30); // addr[56:32], type=2
  i32x8 g1;
  g1[0] = (1 << 16)      // data_size = 2 bytes
        | (1 << 20)      // pad_enable
        | (3 << 22)      // pad_interval: 16 DWORDs (row = 64B)
        | (3 << 25);     // pad_amount:   4 DWORDs (16B -> 80B pitch)
  g1[1] = (int)(((unsigned)pitchElems & 0xFFFFu) << 16);    // tensor_dim0 lo16
  g1[2] = (int)(((unsigned)pitchElems >> 16));              // dim0 hi16, dim1 lo16=0
  g1[3] = (int)(0x10u | (32u << 16));                       // dim1=1M (hi16), tile_dim0=32
  g1[4] = nRows;                                            // tile_dim1
  g1[5] = pitchElems;                                       // tensor_dim0_stride lo32
  g1[6] = 0;
  g1[7] = 0;
  i32x4 z4 = {0, 0, 0, 0};
  i32x8 z8 = {0, 0, 0, 0, 0, 0, 0, 0};
  __builtin_amdgcn_tensor_load_to_lds(g0, g1, z4, z4, z8, 0);
}
#endif

// =====================================================================
// Weight transpose + fp32->bf16 convert: src [K][N] row-major (per batch)
// -> dst [Nr][Kpad] bf16, zero padded (Nr rounded up to N-tile).
// =====================================================================
__global__ __launch_bounds__(256)
void udr_convt_kernel(__bf16* __restrict__ dst, const float* __restrict__ src,
                      int K, int N, int Kpad, int Nr,
                      long long dstBatch, long long srcBatch, int nb)
{
  long long per = (long long)Nr * Kpad;
  long long total = per * nb;
  for (long long i = blockIdx.x * 256ll + threadIdx.x; i < total;
       i += (long long)gridDim.x * 256ll) {
    int bz = (int)(i / per);
    long long r = i % per;
    int n = (int)(r / Kpad);
    int k = (int)(r % Kpad);
    float v = (n < N && k < K) ? src[bz * srcBatch + (long long)k * N + n] : 0.f;
    dst[bz * dstBatch + (long long)n * Kpad + k] = f2bf(v);
  }
}

// =====================================================================
// Batched WMMA GEMM, all-bf16 operands, double-buffered TDM pipeline.
//   acc = A[M,K]bf16 (row pitch lda, K-pad zeroed) x Wt[n][k] (pitch wpitch)
// Epilogue modes:
//   v = acc + bias[n];  if relu: v=max(v,0)
//   if mulX0:  v = mulX0[m,n]*v + addXp[m,n]      (CrossNet combine)
//   if Cf:     Cf[m*ldc+n]  = v                    (fp32 out)
//   if Cb:     Cb[m*ldcb+n] = bf16(v); pad cols [N,npadb) zeroed
// M multiple of 128; tiles 128x64, K-step 32.
// =====================================================================
#define GTM 128
#define GTN 64
#define GTK 32
#define LDSK (GTK + 8)     // 40 bf16 = 80B row pitch

__global__ __launch_bounds__(256)
void udr_gemm_bf16_kernel(const __bf16* __restrict__ A, long long aBatch, int lda,
                          const __bf16* __restrict__ Wt, long long wBatch, int wpitch,
                          float* __restrict__ Cf, long long cBatch, int ldc,
                          __bf16* __restrict__ Cb, int ldcb, int npadb,
                          const float* __restrict__ bias, int biasStride, int relu,
                          const float* __restrict__ mulX0,
                          const float* __restrict__ addXp, int ldx,
                          int M, int N, int K)
{
  __shared__ __bf16 sA[2][GTM][LDSK];   // 20 KB
  __shared__ __bf16 sB[2][GTN][LDSK];   // 10 KB

  const int tid  = threadIdx.x;
  const int lane = tid & 31;
  const int wave = tid >> 5;
  const int bz   = blockIdx.z;

  A  += (long long)bz * aBatch;
  Wt += (long long)bz * wBatch;
  if (Cf) Cf += (long long)bz * cBatch;
  const float* biasp = bias ? (bias + (long long)bz * biasStride) : nullptr;

  const int n0    = blockIdx.x * GTN;
  const int m0blk = blockIdx.y * GTM;
  const int mw    = wave * 16;
  const int ntk   = (K + GTK - 1) / GTK;

  const __bf16* Abase = A  + (long long)m0blk * lda;
  const __bf16* Wbase = Wt + (long long)n0 * wpitch;

  v8f acc[4] = {};

#if USE_TDM
  // prologue: tile 0 into buffer 0 (wave 0: B, wave 1: A)
  if (wave == 0)
    tdm_tile_2d(Wbase, (unsigned)(unsigned long long)(void*)&sB[0][0][0], GTN, wpitch);
  else if (wave == 1)
    tdm_tile_2d(Abase, (unsigned)(unsigned long long)(void*)&sA[0][0][0], GTM, lda);
#endif

  for (int kt = 0; kt < ntk; ++kt) {
    const int cur = kt & 1;
    __syncthreads();   // prior reads of the 'next' buffer complete

#if USE_TDM
    if (kt + 1 < ntk) {
      const int k1 = (kt + 1) * GTK;
      if (wave == 0)
        tdm_tile_2d(Wbase + k1,
                    (unsigned)(unsigned long long)(void*)&sB[cur ^ 1][0][0],
                    GTN, wpitch);
      else if (wave == 1)
        tdm_tile_2d(Abase + k1,
                    (unsigned)(unsigned long long)(void*)&sA[cur ^ 1][0][0],
                    GTM, lda);
      else if (wave == 2 && lane == 0 && kt + 2 < ntk)
        __builtin_prefetch(Abase + (kt + 2) * GTK, 0, 1);
      __builtin_amdgcn_s_wait_tensorcnt(1);   // oldest (tile kt) has landed
    } else {
      __builtin_amdgcn_s_wait_tensorcnt(0);   // last tile: drain
    }
#else
    {
      const int k0 = kt * GTK;
      {   // manual B staging: one uint4 (8 bf16) per thread
        int row = tid >> 2;
        int kh  = (tid & 3) * 8;
        *(uint4*)&sB[cur][row][kh] =
            *(const uint4*)(Wbase + (long long)row * wpitch + k0 + kh);
      }
      {   // manual A staging: two uint4 (16 bf16) per thread
        int row = tid >> 1;
        int kh  = (tid & 1) * 16;
        const uint4* src = (const uint4*)(Abase + (long long)row * lda + k0 + kh);
        uint4* d = (uint4*)&sA[cur][row][kh];
        d[0] = src[0];
        d[1] = src[1];
      }
    }
#endif
    __syncthreads();

    // ---- A fragment: 16x32 bf16, documented wave32 layout ----
    v16bf afrag;
    {
      int r  = mw + (lane & 15);
      int kb = (lane < 16) ? 0 : 8;
#pragma unroll
      for (int i = 0; i < 8; ++i) {
        afrag[i]     = sA[cur][r][kb + i];
        afrag[i + 8] = sA[cur][r][kb + 16 + i];
      }
    }
    // ---- load all 4 B fragments, then 4 WMMAs ----
    v16bf bfrag[4];
#pragma unroll
    for (int j = 0; j < 4; ++j) {
      int col = j * 16 + (lane & 15);
      int kb  = (lane < 16) ? 0 : 8;
#pragma unroll
      for (int i = 0; i < 16; ++i) bfrag[j][i] = sB[cur][col][kb + i + ((i >> 3) << 3)];
    }
#pragma unroll
    for (int j = 0; j < 4; ++j)
      acc[j] = __builtin_amdgcn_wmma_f32_16x16x32_bf16(
          false, afrag, false, bfrag[j], (short)0, acc[j], false, false);
  }

  // ---- epilogue ----
#pragma unroll
  for (int j = 0; j < 4; ++j) {
    int n = n0 + j * 16 + (lane & 15);
    if (n < N) {
      float bv = biasp ? biasp[n] : 0.f;
#pragma unroll
      for (int i = 0; i < 8; ++i) {
        int m = m0blk + mw + ((lane < 16) ? i : 8 + i);
        float v = acc[j][i] + bv;
        if (relu) v = fmaxf(v, 0.f);
        if (mulX0) {
          long long xi = (long long)m * ldx + n;
          v = mulX0[xi] * v + addXp[xi];
        }
        if (Cf) Cf[(long long)m * ldc + n] = v;
        if (Cb) Cb[(long long)m * ldcb + n] = f2bf(v);
      }
    } else if (Cb && n < npadb) {   // zero K-padding columns for consumer GEMM
#pragma unroll
      for (int i = 0; i < 8; ++i) {
        int m = m0blk + mw + ((lane < 16) ? i : 8 + i);
        Cb[(long long)m * ldcb + n] = f2bf(0.f);
      }
    }
  }
}

// =====================================================================
// Embedding gather + DIN attention + x0 assembly (fp32 + padded bf16).
// One block per sample.
// =====================================================================
__global__ __launch_bounds__(256)
void udr_embed_din_kernel(const int* __restrict__ userId, const int* __restrict__ movieId,
                          const int* __restrict__ genre, const int* __restrict__ age,
                          const int* __restrict__ occ, const int* __restrict__ seq,
                          const float* __restrict__ dense,
                          const float* __restrict__ E_user, const float* __restrict__ E_movie,
                          const float* __restrict__ E_genre, const float* __restrict__ E_age,
                          const float* __restrict__ E_occ,
                          const __bf16* __restrict__ W1t,   // [64][512] bf16
                          const float* __restrict__ b1, const float* __restrict__ W2,
                          const float* __restrict__ b2,
                          float* __restrict__ x0, __bf16* __restrict__ x0b)
{
  __shared__ float  s_tgt[DIM];
  __shared__ int    s_idx[SEQT];
  __shared__ float  s_hist[SEQT][DIM];          // 32 KB
  __shared__ __bf16 s_attin[SEQT][4 * DIM + 8]; // 65 KB, stride 520 (16B aligned)
  __shared__ float  s_h[SEQT][ATTH + 8];        // 18 KB
  __shared__ float  s_w[SEQT];

  const int b    = blockIdx.x;
  const int tid  = threadIdx.x;
  const int lane = tid & 31;
  const int wave = tid >> 5;

  const int mid = movieId[b];
  if (tid < SEQT) s_idx[tid] = seq[(long long)b * SEQT + tid];
  for (int d = tid; d < DIM; d += 256)
    s_tgt[d] = (mid != 0) ? E_movie[(long long)mid * DIM + d] : 0.f;
  __syncthreads();

  // gather history + build att_in = [q, k, q-k, q*k] bf16 (2 cols/iter)
  for (int i = tid; i < SEQT * (DIM / 2); i += 256) {
    int t = i >> 6, d = (i & 63) * 2;
    int idx = s_idx[t];
    float2 hv;
    if (idx != 0) hv = *(const float2*)(E_movie + (long long)idx * DIM + d);
    else { hv.x = 0.f; hv.y = 0.f; }
    *(float2*)&s_hist[t][d] = hv;
    float q0 = s_tgt[d], q1 = s_tgt[d + 1];
    unsigned* at = (unsigned*)&s_attin[t][0];
    at[(d) >> 1]            = packbf(q0, q1);
    at[(DIM + d) >> 1]      = packbf(hv.x, hv.y);
    at[(2 * DIM + d) >> 1]  = packbf(q0 - hv.x, q1 - hv.y);
    at[(3 * DIM + d) >> 1]  = packbf(q0 * hv.x, q1 * hv.y);
  }
  __syncthreads();

  // [64x512] @ [512x64] WMMA; 8 waves, 2 N-tiles each
  v8f acc[2] = {};
  const int mrow  = (wave & 3) * 16 + (lane & 15);
  const int nbase = (wave >> 2) * 32;
  for (int k0 = 0; k0 < 4 * DIM; k0 += 32) {
    v16bf afrag;
    int kb = k0 + ((lane < 16) ? 0 : 8);
#pragma unroll
    for (int i = 0; i < 8; ++i) {
      afrag[i]     = s_attin[mrow][kb + i];
      afrag[i + 8] = s_attin[mrow][kb + 16 + i];
    }
    v16bf bfrag[2];
#pragma unroll
    for (int j = 0; j < 2; ++j) {
      int col = nbase + j * 16 + (lane & 15);
      int kk  = k0 + ((lane < 16) ? 0 : 8);
      const __bf16* wp = W1t + (long long)col * (4 * DIM) + kk;
#pragma unroll
      for (int i = 0; i < 16; ++i) bfrag[j][i] = wp[i + ((i >> 3) << 3)];
    }
#pragma unroll
    for (int j = 0; j < 2; ++j)
      acc[j] = __builtin_amdgcn_wmma_f32_16x16x32_bf16(
          false, afrag, false, bfrag[j], (short)0, acc[j], false, false);
  }
#pragma unroll
  for (int j = 0; j < 2; ++j) {
    int n = nbase + j * 16 + (lane & 15);
    float bv = b1[n];
#pragma unroll
    for (int i = 0; i < 8; ++i) {
      int m = (wave & 3) * 16 + ((lane < 16) ? i : 8 + i);
      s_h[m][n] = fmaxf(acc[j][i] + bv, 0.f);
    }
  }
  __syncthreads();

  // scores + mask
  if (tid < SEQT) {
    float s = b2[0];
    for (int hh = 0; hh < ATTH; ++hh) s += s_h[tid][hh] * W2[hh];
    s_w[tid] = (s_idx[tid] != 0) ? s : -1e9f;
  }
  __syncthreads();
  if (tid == 0) {
    float mx = -1e30f;
    for (int t = 0; t < SEQT; ++t) mx = fmaxf(mx, s_w[t]);
    float sum = 0.f;
    for (int t = 0; t < SEQT; ++t) { float e = __expf(s_w[t] - mx); s_w[t] = e; sum += e; }
    float inv = 1.f / sum;
    for (int t = 0; t < SEQT; ++t) s_w[t] *= inv;
  }
  __syncthreads();

  float*  xrow = x0  + (long long)b * TOTF;
  __bf16* xbrw = x0b + (long long)b * TOTP;
  // interest = sum_t w_t * hist[t]
  if (tid < DIM) {
    float ai = 0.f;
    for (int t = 0; t < SEQT; ++t) ai += s_w[t] * s_hist[t][tid];
    xrow[5 * DIM + tid] = ai;
    xbrw[5 * DIM + tid] = f2bf(ai);
  }
  // sparse embeddings + dense (+ K-pad zeros in bf16 copy)
  const int uid = userId[b], gid = genre[b], aid = age[b], oid = occ[b];
  for (int c = tid; c < TOTF; c += 256) {
    float v;
    if      (c < 128) v = (uid != 0) ? E_user [(long long)uid * DIM + c]        : 0.f;
    else if (c < 256) v = (mid != 0) ? E_movie[(long long)mid * DIM + (c-128)]  : 0.f;
    else if (c < 384) v = (gid != 0) ? E_genre[(long long)gid * DIM + (c-256)]  : 0.f;
    else if (c < 512) v = (aid != 0) ? E_age  [(long long)aid * DIM + (c-384)]  : 0.f;
    else if (c < 640) v = (oid != 0) ? E_occ  [(long long)oid * DIM + (c-512)]  : 0.f;
    else if (c < 768) continue;  // interest written above
    else              v = dense[(long long)b * 16 + (c - 768)];
    xrow[c] = v;
    xbrw[c] = f2bf(v);
  }
  if (tid < TOTP - TOTF) xbrw[TOTF + tid] = f2bf(0.f);
}

// =====================================================================
// Gates + expert mix + heads. One wave (32 lanes) per sample.
// =====================================================================
__global__ __launch_bounds__(256)
void udr_moe_head_kernel(const float* __restrict__ xc, const float* __restrict__ eout,
                         const float* __restrict__ gate_W, const float* __restrict__ gate_b,
                         const float* __restrict__ head_W, const float* __restrict__ head_b,
                         float* __restrict__ out, int Bsz)
{
  const int wave = threadIdx.x >> 5;
  const int lane = threadIdx.x & 31;
  const int b = blockIdx.x * 8 + wave;
  if (b >= Bsz) return;

  const float* xr = xc + (long long)b * TOTF;
  float g[8] = {0.f, 0.f, 0.f, 0.f, 0.f, 0.f, 0.f, 0.f};
  for (int d = lane; d < TOTF; d += 32) {
    float xv = xr[d];
#pragma unroll
    for (int t = 0; t < 2; ++t)
#pragma unroll
      for (int e = 0; e < NEXP; ++e)
        g[t * 4 + e] += xv * gate_W[((long long)t * TOTF + d) * NEXP + e];
  }
#pragma unroll
  for (int off = 16; off; off >>= 1)
#pragma unroll
    for (int i = 0; i < 8; ++i) g[i] += __shfl_xor(g[i], off, 32);

  float gt[2][4];
#pragma unroll
  for (int t = 0; t < 2; ++t) {
    float mx = -1e30f;
#pragma unroll
    for (int e = 0; e < 4; ++e) { gt[t][e] = g[t * 4 + e] + gate_b[t * 4 + e]; mx = fmaxf(mx, gt[t][e]); }
    float s = 0.f;
#pragma unroll
    for (int e = 0; e < 4; ++e) { gt[t][e] = __expf(gt[t][e] - mx); s += gt[t][e]; }
    float inv = 1.f / s;
#pragma unroll
    for (int e = 0; e < 4; ++e) gt[t][e] *= inv;
  }

  float lg0 = 0.f, lg1 = 0.f;
  for (int o = lane; o < 128; o += 32) {
    float m0 = 0.f, m1 = 0.f;
#pragma unroll
    for (int e = 0; e < 4; ++e) {
      float ev = eout[((long long)e * Bsz + b) * 128 + o];
      m0 += gt[0][e] * ev; m1 += gt[1][e] * ev;
    }
    lg0 += m0 * head_W[o];
    lg1 += m1 * head_W[128 + o];
  }
#pragma unroll
  for (int off = 16; off; off >>= 1) {
    lg0 += __shfl_xor(lg0, off, 32);
    lg1 += __shfl_xor(lg1, off, 32);
  }
  if (lane == 0) {
    out[b]                  = lg0 + head_b[0];
    out[(long long)Bsz + b] = lg1 + head_b[1];
  }
}

// =====================================================================
extern "C" void kernel_launch(void* const* d_in, const int* in_sizes, int n_in,
                              void* d_out, int out_size, void* d_ws, size_t ws_size,
                              hipStream_t stream)
{
  const int*   userId  = (const int*)  d_in[0];
  const int*   movieId = (const int*)  d_in[1];
  const int*   genreI  = (const int*)  d_in[2];
  const int*   ageI    = (const int*)  d_in[3];
  const int*   occI    = (const int*)  d_in[4];
  const int*   seq     = (const int*)  d_in[5];
  const float* dense   = (const float*)d_in[6];
  const float* E_user  = (const float*)d_in[7];
  const float* E_movie = (const float*)d_in[8];
  const float* E_genre = (const float*)d_in[9];
  const float* E_age   = (const float*)d_in[10];
  const float* E_occ   = (const float*)d_in[11];
  const float* att_W1  = (const float*)d_in[12];
  const float* att_b1  = (const float*)d_in[13];
  const float* att_W2  = (const float*)d_in[14];
  const float* att_b2  = (const float*)d_in[15];
  const float* cross_W = (const float*)d_in[16];
  const float* cross_b = (const float*)d_in[17];
  const float* exp_W1  = (const float*)d_in[18];
  const float* exp_b1  = (const float*)d_in[19];
  const float* exp_W2  = (const float*)d_in[20];
  const float* exp_b2  = (const float*)d_in[21];
  const float* gate_W  = (const float*)d_in[22];
  const float* gate_b  = (const float*)d_in[23];
  const float* head_W  = (const float*)d_in[24];
  const float* head_b  = (const float*)d_in[25];
  float* out = (float*)d_out;

  // ---- workspace bump allocator ----
  char* wp = (char*)d_ws;
  auto walloc = [&](size_t bytes) -> void* {
    void* r = (void*)wp;
    wp += (bytes + 255) & ~(size_t)255;
    return r;
  };
  float*  x0      = (float*) walloc((size_t)BATCH * TOTF * 4);       // 51.4 MB
  __bf16* x0b     = (__bf16*)walloc((size_t)BATCH * TOTP * 2);       // 26.2 MB
  float*  xc      = (float*) walloc((size_t)BATCH * TOTF * 4);       // 51.4 MB
  __bf16* xb0     = (__bf16*)walloc((size_t)BATCH * TOTP * 2);       // 26.2 MB
  __bf16* xb1     = (__bf16*)walloc((size_t)BATCH * TOTP * 2);       // 26.2 MB
  __bf16* ehb     = (__bf16*)walloc((size_t)BATCH * 1024 * 2);       // 33.5 MB
  float*  eout    = (float*) walloc((size_t)NEXP * BATCH * 128 * 4); // 33.6 MB
  __bf16* W1t     = (__bf16*)walloc((size_t)64 * 512 * 2);
  __bf16* crossWt = (__bf16*)walloc((size_t)3 * 832 * 800 * 2);
  __bf16* expW1t  = (__bf16*)walloc((size_t)1024 * 800 * 2);
  __bf16* expW2t  = (__bf16*)walloc((size_t)NEXP * 128 * 256 * 2);

  auto cblk = [](long long n) { return (unsigned)((n + 255) / 256); };

  // ---- weight conversions (deterministic, every call) ----
  udr_convt_kernel<<<cblk((long long)64 * 512), 256, 0, stream>>>(
      W1t, att_W1, 512, 64, 512, 64, 0, 0, 1);
  udr_convt_kernel<<<cblk(3ll * 832 * 800), 256, 0, stream>>>(
      crossWt, cross_W, TOTF, TOTF, 800, 832,
      832ll * 800, (long long)TOTF * TOTF, 3);
  udr_convt_kernel<<<cblk(4ll * 256 * 800), 256, 0, stream>>>(
      expW1t, exp_W1, TOTF, 256, 800, 256,
      256ll * 800, (long long)TOTF * 256, 4);
  udr_convt_kernel<<<cblk(4ll * 128 * 256), 256, 0, stream>>>(
      expW2t, exp_W2, 256, 128, 256, 128,
      128ll * 256, 256ll * 128, 4);

  // ---- embeddings + DIN attention -> x0 (fp32) + x0b (bf16, padded) ----
  udr_embed_din_kernel<<<BATCH, 256, 0, stream>>>(
      userId, movieId, genreI, ageI, occI, seq, dense,
      E_user, E_movie, E_genre, E_age, E_occ,
      W1t, att_b1, att_W2, att_b2, x0, x0b);

  // ---- CrossNetV2: 3 layers, combine fused into GEMM epilogue ----
  dim3 gcross((TOTF + GTN - 1) / GTN, BATCH / GTM, 1);
  {
    const __bf16* ain[3]  = {x0b, xb0, xb1};
    __bf16*       bout[3] = {xb0, xb1, xb0};
    for (int l = 0; l < 3; ++l) {
      const float* xprev = (l == 0) ? x0 : xc;
      udr_gemm_bf16_kernel<<<gcross, 256, 0, stream>>>(
          ain[l], 0, TOTP,
          crossWt + (long long)l * 832 * 800, 0, 800,
          xc, 0, TOTF,
          bout[l], TOTP, TOTP,
          cross_b + (long long)l * TOTF, 0, 0,
          x0, xprev, TOTF,
          BATCH, TOTF, TOTF);
    }
  }
  __bf16* xfin = xb0;   // bf16 x after layer 2

  // ---- MMoE expert layer 1 (4 experts fused: N=1024), ReLU, bf16 out ----
  dim3 gexp1((1024 + GTN - 1) / GTN, BATCH / GTM, 1);
  udr_gemm_bf16_kernel<<<gexp1, 256, 0, stream>>>(
      xfin, 0, TOTP,
      expW1t, 0, 800,
      nullptr, 0, 0,
      ehb, 1024, 1024,
      exp_b1, 0, 1,
      nullptr, nullptr, 0,
      BATCH, 1024, TOTF);

  // ---- MMoE expert layer 2 (batched over 4 experts), fp32 out ----
  dim3 gexp2((128 + GTN - 1) / GTN, BATCH / GTM, NEXP);
  udr_gemm_bf16_kernel<<<gexp2, 256, 0, stream>>>(
      ehb, 256, 1024,
      expW2t, 128ll * 256, 256,
      eout, (long long)BATCH * 128, 128,
      nullptr, 0, 0,
      exp_b2, 128, 0,
      nullptr, nullptr, 0,
      BATCH, 128, 256);

  // ---- gates + mix + heads ----
  udr_moe_head_kernel<<<BATCH / 8, 256, 0, stream>>>(
      xc, eout, gate_W, gate_b, head_W, head_b, out, BATCH);

  (void)in_sizes; (void)n_in; (void)out_size; (void)ws_size;
}
